// MedSAMIBot_20615843021506
// MI455X (gfx1250) — compile-verified, matrix operand order checked by reading, not run on hardware
//
#include <hip/hip_runtime.h>

typedef __bf16 bf16;
typedef __attribute__((ext_vector_type(16))) __bf16 v16bf;
typedef __attribute__((ext_vector_type(8)))  float  v8f;
typedef __attribute__((ext_vector_type(8)))  unsigned int v8u;

#define EMB   768
#define HEADS 12
#define HD    64
#define WS    14
#define MLPD  3072
#define BATCH 4
#define HW    4096
#define NTOK  4097          // tokens per batch (cls + 64*64)
#define MTOK  (BATCH*NTOK)  // 16388
#define NWIN  25            // windows per batch (5x5 over padded 70x70)
#define NW    (BATCH*NWIN)  // 100
#define TREAL 197           // real tokens per window
#define TP    208           // padded tokens per window (13*16)
#define WHN   (NW*HEADS)    // 1200
#define ASCALE 0.125f       // HD^-0.5

static __device__ __forceinline__ v8f wmma_bf16(v16bf a, v16bf b, v8f c) {
  return __builtin_amdgcn_wmma_f32_16x16x32_bf16(false, a, false, b, (short)0, c, false, false);
}
static __device__ __forceinline__ v8f zero8() {
  v8f z = {0.f,0.f,0.f,0.f,0.f,0.f,0.f,0.f};
  return z;
}

// Load a 16x32 bf16 WMMA fragment (A layout; B layout is the mirrored
// N-major load, so the same code serves both when W is stored K-major).
// Per ISA 7.12.2: lane l holds row (l&15); dword v holds K pair
//   K = (l<16?0:8) + (v<4 ? 2v : 16+2(v-4))  =>  dword index = (l<16?0:4) + (v<4 ? v : 4+v)
static __device__ __forceinline__ v16bf load_frag(const bf16* base, int stride,
                                                  int row0, int kbase, int lane) {
  const unsigned int* p =
      (const unsigned int*)(base + (size_t)(row0 + (lane & 15)) * stride + kbase);
  int h = (lane < 16) ? 0 : 4;
  v8u u;
#pragma unroll
  for (int v = 0; v < 8; ++v) {
    int idx = h + ((v < 4) ? v : (4 + v));
    u[v] = p[idx];
  }
  return __builtin_bit_cast(v16bf, u);
}

// ---------------------------------------------------------------- weights
__global__ __launch_bounds__(256) void k_wt(const float* W, bf16* Wt, int K, int N) {
  long idx = (long)blockIdx.x * 256 + threadIdx.x;
  if (idx >= (long)K * N) return;
  int k = (int)(idx / N), n = (int)(idx % N);
  Wt[(size_t)n * K + k] = (bf16)W[idx];
}

// ---------------------------------------------------------------- layernorm
__global__ __launch_bounds__(256) void k_layernorm(const float* tok, const float* x,
                                                   const float* cls, const float* s,
                                                   const float* bta, bf16* out) {
  int row = blockIdx.x;            // 0..MTOK-1
  int b = row / NTOK, t = row % NTOK;
  const float* src;
  if (tok) src = tok + (size_t)row * EMB;
  else src = (t == 0) ? (cls + (size_t)b * EMB)
                      : (x + ((size_t)b * HW + (t - 1)) * EMB);
  float v[3], lsum = 0.f, lsq = 0.f;
#pragma unroll
  for (int i = 0; i < 3; ++i) {
    float f = src[threadIdx.x + i * 256];
    v[i] = f; lsum += f; lsq += f * f;
  }
  for (int off = 16; off; off >>= 1) {
    lsum += __shfl_xor(lsum, off, 32);
    lsq  += __shfl_xor(lsq,  off, 32);
  }
  __shared__ float rs[8], rq[8];
  int w = threadIdx.x >> 5, l = threadIdx.x & 31;
  if (l == 0) { rs[w] = lsum; rq[w] = lsq; }
  __syncthreads();
  if (w == 0) {
    float a = (l < 8) ? rs[l] : 0.f, c = (l < 8) ? rq[l] : 0.f;
    for (int off = 4; off; off >>= 1) { a += __shfl_xor(a, off, 32); c += __shfl_xor(c, off, 32); }
    if (l == 0) { rs[0] = a; rq[0] = c; }
  }
  __syncthreads();
  float mu  = rs[0] * (1.f / EMB);
  float var = rq[0] * (1.f / EMB) - mu * mu;
  float inv = rsqrtf(var + 1e-6f);
#pragma unroll
  for (int i = 0; i < 3; ++i) {
    int e = threadIdx.x + i * 256;
    out[(size_t)row * EMB + e] = (bf16)((v[i] - mu) * inv * s[e] + bta[e]);
  }
}

// ---------------------------------------------------------------- window gather
__global__ __launch_bounds__(256) void k_window(const bf16* tn, bf16* xw) {
  int row = blockIdx.x;            // 0..NW*TP-1
  int w = row / TP, t = row % TP;
  int b = w / NWIN, wi = w % NWIN;
  int wy = wi / 5, wx = wi % 5;
  const bf16* src = nullptr;
  if (t == 0) src = tn + (size_t)(b * NTOK) * EMB;
  else if (t < TREAL) {
    int q = t - 1, r = wy * WS + q / WS, c = wx * WS + q % WS;
    if (r < 64 && c < 64) src = tn + ((size_t)b * NTOK + 1 + r * 64 + c) * EMB;
  }
  bf16* dst = xw + (size_t)row * EMB;
  for (int e = threadIdx.x; e < EMB; e += 256)
    dst[e] = src ? src[e] : (bf16)0.f;
}

// ---------------------------------------------------------------- generic GEMM
// C = A(MxK bf16, row-major) x W^T(NxK bf16, row-major) + epilogue(MODE)
struct GemmP {
  const bf16* A; const bf16* W; const float* bias;
  float* outf; bf16* outb; bf16* q; bf16* k; bf16* v;
  int M, N, K;
};
#define BM 128
#define BN 128
#define BK 32
#define LDT 34   // LDS row stride (elems), 17 dwords -> conflict-friendly

template<int MODE>
__global__ __launch_bounds__(256) void k_gemm(GemmP p) {
  __shared__ bf16 As[BM][LDT];
  __shared__ bf16 Bs[BN][LDT];
  int bn0 = blockIdx.x * BN, bm0 = blockIdx.y * BM;
  int tid = threadIdx.x, lane = tid & 31, wave = tid >> 5;
  int wm = wave >> 2, wn = wave & 3;   // 2x4 wave grid -> 64x32 per wave
  v8f acc[4][2];
#pragma unroll
  for (int mi = 0; mi < 4; ++mi)
#pragma unroll
    for (int ni = 0; ni < 2; ++ni) acc[mi][ni] = zero8();

  int arow = bm0 + (tid >> 1); if (arow >= p.M) arow = p.M - 1;
  int brow = bn0 + (tid >> 1); if (brow >= p.N) brow = p.N - 1;
  int coff = (tid & 1) * 16;           // element offset within 32-wide slab

  for (int k0 = 0; k0 < p.K; k0 += BK) {
    const uint4* ga = (const uint4*)(p.A + (size_t)arow * p.K + k0 + coff);
    const uint4* gb = (const uint4*)(p.W + (size_t)brow * p.K + k0 + coff);
    uint4 a0 = ga[0], a1 = ga[1];
    uint4 b0 = gb[0], b1 = gb[1];
    if (k0 + BK < p.K) {
      __builtin_prefetch((const void*)(p.A + (size_t)arow * p.K + k0 + BK + coff), 0, 1);
      __builtin_prefetch((const void*)(p.W + (size_t)brow * p.K + k0 + BK + coff), 0, 1);
    }
    __syncthreads();
    unsigned int* da = (unsigned int*)&As[tid >> 1][coff];
    da[0]=a0.x; da[1]=a0.y; da[2]=a0.z; da[3]=a0.w;
    da[4]=a1.x; da[5]=a1.y; da[6]=a1.z; da[7]=a1.w;
    unsigned int* db = (unsigned int*)&Bs[tid >> 1][coff];
    db[0]=b0.x; db[1]=b0.y; db[2]=b0.z; db[3]=b0.w;
    db[4]=b1.x; db[5]=b1.y; db[6]=b1.z; db[7]=b1.w;
    __syncthreads();

    v16bf af[4], bfg[2];
#pragma unroll
    for (int mi = 0; mi < 4; ++mi)
      af[mi] = load_frag(&As[0][0], LDT, wm * 64 + mi * 16, 0, lane);
#pragma unroll
    for (int ni = 0; ni < 2; ++ni)
      bfg[ni] = load_frag(&Bs[0][0], LDT, wn * 32 + ni * 16, 0, lane);
#pragma unroll
    for (int mi = 0; mi < 4; ++mi)
#pragma unroll
      for (int ni = 0; ni < 2; ++ni)
        acc[mi][ni] = wmma_bf16(af[mi], bfg[ni], acc[mi][ni]);
  }

  int rplus = (lane >= 16) ? 8 : 0;
  int col = lane & 15;
#pragma unroll
  for (int mi = 0; mi < 4; ++mi)
#pragma unroll
    for (int ni = 0; ni < 2; ++ni)
#pragma unroll
      for (int r = 0; r < 8; ++r) {
        int m = bm0 + wm * 64 + mi * 16 + r + rplus;
        int n = bn0 + wn * 32 + ni * 16 + col;
        if (m >= p.M) continue;
        float val = acc[mi][ni][r] + p.bias[n];
        if constexpr (MODE == 0) {          // QKV scatter -> head-major bf16
          int part = n / EMB, hh = (n % EMB) / HD, d = n % HD;
          int w = m / TP, t = m % TP;
          size_t idx = (((size_t)w * HEADS + hh) * TP + t) * HD + d;
          bf16 bv = (bf16)val;
          if (part == 0) p.q[idx] = bv;
          else if (part == 1) p.k[idx] = bv;
          else p.v[idx] = bv;
        } else if constexpr (MODE == 1) {   // proj -> f32
          p.outf[(size_t)m * p.N + n] = val;
        } else if constexpr (MODE == 2) {   // fc1 -> gelu -> bf16
          float g = 0.5f * val *
                    (1.f + tanhf(0.7978845608f * (val + 0.044715f * val * val * val)));
          p.outb[(size_t)m * p.N + n] = (bf16)g;
        } else {                            // fc2 -> residual add
          p.outf[(size_t)m * p.N + n] += val;
        }
      }
}

// ---------------------------------------------------------------- rel-pos dots
__global__ __launch_bounds__(32) void k_relpos(const bf16* Q, const float* rph,
                                               const float* rpw, float* relh, float* relw) {
  int wh = blockIdx.x, q = blockIdx.y;     // wh<1200, q<196
  int i = q / WS, j = q % WS;
  int o = threadIdx.x;
  if (o >= 2 * WS) return;
  const bf16* qv = Q + ((size_t)wh * TP + 1 + q) * HD;
  const float* rv = (o < WS) ? (rph + (size_t)(i - o + WS - 1) * HD)
                             : (rpw + (size_t)(j - (o - WS) + WS - 1) * HD);
  float acc = 0.f;
  for (int c = 0; c < HD; ++c) acc += (float)qv[c] * rv[c];
  if (o < WS) relh[((size_t)wh * 196 + q) * WS + o] = acc;
  else        relw[((size_t)wh * 196 + q) * WS + (o - WS)] = acc;
}

// ---------------------------------------------------------------- attention
__global__ __launch_bounds__(128) void k_attn(const bf16* Qg, const bf16* Kg, const bf16* Vg,
                                              const float* relh, const float* relw, bf16* O) {
  __shared__ bf16 Vt[HD][224];        // V transposed, padded to 224 tokens
  __shared__ bf16 Sb[4][16 * 224];    // per-wave probability tile (A layout source)
  int wh = blockIdx.x;
  int w = wh / HEADS, h = wh % HEADS;
  int tid = threadIdx.x, lane = tid & 31, wave = tid >> 5;
  const bf16* Qb = Qg + (size_t)wh * TP * HD;
  const bf16* Kb = Kg + (size_t)wh * TP * HD;
  const bf16* Vb = Vg + (size_t)wh * TP * HD;

  for (int idx = tid; idx < HD * 224; idx += 128) {
    int d = idx / 224, t = idx % 224;
    Vt[d][t] = (t < TP) ? Vb[(size_t)t * HD + d] : (bf16)0.f;
  }
  __syncthreads();

  int rplus = (lane >= 16) ? 8 : 0;
  int col = lane & 15;
  bf16* Sw = Sb[wave];

  for (int qt = wave; qt < 13; qt += 4) {
    v16bf qa0 = load_frag(Qb, HD, qt * 16, 0, lane);
    v16bf qa1 = load_frag(Qb, HD, qt * 16, 32, lane);
    v8f acc[13];
#pragma unroll
    for (int kt = 0; kt < 13; ++kt) acc[kt] = zero8();
    for (int kt = 0; kt < 13; ++kt) {
      v16bf kb0 = load_frag(Kb, HD, kt * 16, 0, lane);
      v16bf kb1 = load_frag(Kb, HD, kt * 16, 32, lane);
      acc[kt] = wmma_bf16(qa0, kb0, acc[kt]);
      acc[kt] = wmma_bf16(qa1, kb1, acc[kt]);
    }
    // scale + rel-pos bias + padding mask
    for (int kt = 0; kt < 13; ++kt) {
#pragma unroll
      for (int r = 0; r < 8; ++r) {
        int tq = qt * 16 + r + rplus;
        int tk = kt * 16 + col;
        float vls = acc[kt][r] * ASCALE;
        if (tq >= TREAL)      vls = (tk == 0) ? 0.f : -1e30f;  // dead rows, keep finite
        else if (tk >= TREAL) vls = -1e30f;                    // padded keys
        else if (tk >= 1 && tq >= 1) {
          int qq = tq - 1, kk = tk - 1;
          vls += relh[((size_t)wh * 196 + qq) * WS + (kk / WS)] +
                 relw[((size_t)wh * 196 + qq) * WS + (kk % WS)];
        }
        acc[kt][r] = vls;
      }
    }
    // row softmax (rows split across 16-lane halves, reduce with shuffles)
#pragma unroll
    for (int r = 0; r < 8; ++r) {
      float mx = -1e30f;
      for (int kt = 0; kt < 13; ++kt) mx = fmaxf(mx, acc[kt][r]);
      for (int off = 8; off; off >>= 1) mx = fmaxf(mx, __shfl_xor(mx, off, 32));
      float sum = 0.f;
      for (int kt = 0; kt < 13; ++kt) {
        float e = __expf(acc[kt][r] - mx);
        acc[kt][r] = e; sum += e;
      }
      for (int off = 8; off; off >>= 1) sum += __shfl_xor(sum, off, 32);
      float inv = 1.f / sum;
      bf16* srow = Sw + (size_t)(r + rplus) * 224;
      for (int kt = 0; kt < 13; ++kt)
        srow[kt * 16 + col] = (bf16)(acc[kt][r] * inv);
    }
    if (lane < 16)
      for (int rr = 0; rr < 16; ++rr) Sw[rr * 224 + 208 + lane] = (bf16)0.f;

    // P x V  (contraction over 224 tokens, 7 chunks of 32)
    v8f out[4];
#pragma unroll
    for (int nt = 0; nt < 4; ++nt) out[nt] = zero8();
    for (int kc = 0; kc < 7; ++kc) {
      v16bf sa = load_frag(Sw, 224, 0, kc * 32, lane);
#pragma unroll
      for (int nt = 0; nt < 4; ++nt) {
        v16bf vb = load_frag(&Vt[0][0], 224, nt * 16, kc * 32, lane);
        out[nt] = wmma_bf16(sa, vb, out[nt]);
      }
    }
#pragma unroll
    for (int nt = 0; nt < 4; ++nt)
#pragma unroll
      for (int r = 0; r < 8; ++r) {
        int tq = qt * 16 + r + rplus;
        O[((size_t)w * TP + tq) * EMB + h * HD + nt * 16 + col] = (bf16)out[nt][r];
      }
  }
}

// ---------------------------------------------------------------- merge + residual
__global__ __launch_bounds__(256) void k_combine(const float* x, const float* cls,
                                                 const float* OW, float* out) {
  int row = blockIdx.x;            // 0..MTOK-1
  int b = row / NTOK, t = row % NTOK;
  for (int e = threadIdx.x; e < EMB; e += 256) {
    float sc = (t == 0) ? cls[(size_t)b * EMB + e]
                        : x[((size_t)b * HW + t - 1) * EMB + e];
    float y;
    if (t == 0) {
      float s = 0.f;
      for (int wq = 0; wq < NWIN; ++wq)
        s += OW[((size_t)(b * NWIN + wq) * TP) * EMB + e];
      y = s * (1.f / NWIN);
    } else {
      int q = t - 1, r = q / 64, c = q % 64;
      int w = b * NWIN + (r / WS) * 5 + (c / WS);
      int tw = 1 + (r % WS) * WS + (c % WS);
      y = OW[((size_t)w * TP + tw) * EMB + e];
    }
    out[(size_t)row * EMB + e] = sc + y;
  }
}

// ---------------------------------------------------------------- launch
extern "C" void kernel_launch(void* const* d_in, const int* in_sizes, int n_in,
                              void* d_out, int out_size, void* d_ws, size_t ws_size,
                              hipStream_t stream) {
  (void)in_sizes; (void)n_in; (void)out_size; (void)ws_size;
  const float* x      = (const float*)d_in[0];
  const float* cls    = (const float*)d_in[1];
  const float* qkv_w  = (const float*)d_in[2];
  const float* qkv_b  = (const float*)d_in[3];
  const float* proj_w = (const float*)d_in[4];
  const float* proj_b = (const float*)d_in[5];
  const float* rph    = (const float*)d_in[6];
  const float* rpw    = (const float*)d_in[7];
  const float* n1s    = (const float*)d_in[8];
  const float* n1b    = (const float*)d_in[9];
  const float* n2s    = (const float*)d_in[10];
  const float* n2b    = (const float*)d_in[11];
  const float* fc1_w  = (const float*)d_in[12];
  const float* fc1_b  = (const float*)d_in[13];
  const float* fc2_w  = (const float*)d_in[14];
  const float* fc2_b  = (const float*)d_in[15];
  float* out = (float*)d_out;

  size_t off = 0;
  auto alloc = [&](size_t bytes) {
    void* p = (char*)d_ws + off;
    off += (bytes + 255) & ~(size_t)255;
    return p;
  };
  bf16*  tn     = (bf16*)alloc((size_t)MTOK * EMB * 2);
  bf16*  xw     = (bf16*)alloc((size_t)NW * TP * EMB * 2);
  bf16*  wqkvT  = (bf16*)alloc((size_t)3 * EMB * EMB * 2);
  bf16*  wprojT = (bf16*)alloc((size_t)EMB * EMB * 2);
  bf16*  wfc1T  = (bf16*)alloc((size_t)MLPD * EMB * 2);
  bf16*  wfc2T  = (bf16*)alloc((size_t)EMB * MLPD * 2);
  bf16*  Qb     = (bf16*)alloc((size_t)WHN * TP * HD * 2);
  bf16*  Kb     = (bf16*)alloc((size_t)WHN * TP * HD * 2);
  bf16*  Vb     = (bf16*)alloc((size_t)WHN * TP * HD * 2);
  float* relh   = (float*)alloc((size_t)WHN * 196 * WS * 4);
  float* relw   = (float*)alloc((size_t)WHN * 196 * WS * 4);
  bf16*  Ob     = (bf16*)alloc((size_t)NW * TP * EMB * 2);
  float* OW     = (float*)alloc((size_t)NW * TP * EMB * 4);
  bf16*  h2     = (bf16*)alloc((size_t)MTOK * EMB * 2);
  bf16*  m1     = (bf16*)alloc((size_t)MTOK * MLPD * 2);

  k_wt<<<(unsigned)(((size_t)EMB * 3 * EMB + 255) / 256), 256, 0, stream>>>(qkv_w, wqkvT, EMB, 3 * EMB);
  k_wt<<<(unsigned)(((size_t)EMB * EMB + 255) / 256), 256, 0, stream>>>(proj_w, wprojT, EMB, EMB);
  k_wt<<<(unsigned)(((size_t)EMB * MLPD + 255) / 256), 256, 0, stream>>>(fc1_w, wfc1T, EMB, MLPD);
  k_wt<<<(unsigned)(((size_t)MLPD * EMB + 255) / 256), 256, 0, stream>>>(fc2_w, wfc2T, MLPD, EMB);

  k_layernorm<<<MTOK, 256, 0, stream>>>(nullptr, x, cls, n1s, n1b, tn);
  k_window<<<NW * TP, 256, 0, stream>>>(tn, xw);

  GemmP p0 = {xw, wqkvT, qkv_b, nullptr, nullptr, Qb, Kb, Vb, NW * TP, 3 * EMB, EMB};
  k_gemm<0><<<dim3((3 * EMB) / BN, (NW * TP + BM - 1) / BM), 256, 0, stream>>>(p0);

  k_relpos<<<dim3(WHN, 196), 32, 0, stream>>>(Qb, rph, rpw, relh, relw);
  k_attn<<<WHN, 128, 0, stream>>>(Qb, Kb, Vb, relh, relw, Ob);

  GemmP p1 = {Ob, wprojT, proj_b, OW, nullptr, nullptr, nullptr, nullptr, NW * TP, EMB, EMB};
  k_gemm<1><<<dim3(EMB / BN, (NW * TP + BM - 1) / BM), 256, 0, stream>>>(p1);

  k_combine<<<MTOK, 256, 0, stream>>>(x, cls, OW, out);
  k_layernorm<<<MTOK, 256, 0, stream>>>(out, nullptr, nullptr, n2s, n2b, h2);

  GemmP p2 = {h2, wfc1T, fc1_b, nullptr, m1, nullptr, nullptr, nullptr, MTOK, MLPD, EMB};
  k_gemm<2><<<dim3(MLPD / BN, (MTOK + BM - 1) / BM), 256, 0, stream>>>(p2);

  GemmP p3 = {m1, wfc2T, fc2_b, out, nullptr, nullptr, nullptr, nullptr, MTOK, EMB, MLPD};
  k_gemm<3><<<dim3(EMB / BN, (MTOK + BM - 1) / BM), 256, 0, stream>>>(p3);
}
